// PointNet2ClassPcModel_55173149884754
// MI455X (gfx1250) — matrix operand order repaired and checked by reading, not compiled, not run on hardware
//
#include <hip/hip_runtime.h>
#include <hip/hip_bf16.h>
#include <stdint.h>

typedef __attribute__((ext_vector_type(16))) _Float16 v16h;
typedef __attribute__((ext_vector_type(8)))  _Float16 v8h;
typedef __attribute__((ext_vector_type(8)))  float    v8f;

#define INV_BN 0.99999500003749971f   // 1/sqrt(1+1e-5), eval-mode BN folded into weights

// ---------------------------------------------------------------------------
// Farthest point sampling: one block per point cloud, LDS-resident min-dist.
// ---------------------------------------------------------------------------
__global__ void fps_kernel(const float* __restrict__ pts, int* __restrict__ out_idx,
                           int N, int S) {
  __shared__ float dist[4096];
  __shared__ float sval[256];
  __shared__ int   sidx[256];
  __shared__ int   sfar;
  const int b = blockIdx.x;
  const int tid = threadIdx.x;
  const float* P = pts + (size_t)b * N * 3;
  for (int n = tid; n < N; n += 256) dist[n] = 1e10f;
  if (tid == 0) sfar = 0;
  __syncthreads();
  for (int it = 0; it < S; ++it) {
    int far = sfar;
    if (tid == 0) out_idx[b * S + it] = far;
    float fx = P[far * 3 + 0], fy = P[far * 3 + 1], fz = P[far * 3 + 2];
    float best = -1.0f; int bi = 0;
    for (int n = tid; n < N; n += 256) {
      float dx = P[n * 3 + 0] - fx, dy = P[n * 3 + 1] - fy, dz = P[n * 3 + 2] - fz;
      float d = dx * dx + dy * dy + dz * dz;
      float dm = fminf(dist[n], d);
      dist[n] = dm;
      if (dm > best) { best = dm; bi = n; }
    }
    sval[tid] = best; sidx[tid] = bi;
    __syncthreads();
    for (int s = 128; s > 0; s >>= 1) {
      if (tid < s && sval[tid + s] > sval[tid]) { sval[tid] = sval[tid + s]; sidx[tid] = sidx[tid + s]; }
      __syncthreads();
    }
    if (tid == 0) sfar = sidx[0];
    __syncthreads();
  }
}

// Gather sampled centers: out[b,s,:] = pts[b, idx[b,s], :]
__global__ void gather_xyz_kernel(const float* __restrict__ pts, const int* __restrict__ idx,
                                  float* __restrict__ out, int N, int S, int total) {
  int t = blockIdx.x * blockDim.x + threadIdx.x;
  if (t >= total) return;
  int b = t / S;
  int n = idx[t];
  const float* p = pts + ((size_t)b * N + n) * 3;
  out[(size_t)t * 3 + 0] = p[0];
  out[(size_t)t * 3 + 1] = p[1];
  out[(size_t)t * 3 + 2] = p[2];
}

// Ball query: first K indices (ascending) within radius; backfill with first hit.
__global__ void ballquery_kernel(const float* __restrict__ centers, const float* __restrict__ pts,
                                 int* __restrict__ gi, float r2, int N, int S, int K, int total) {
  int t = blockIdx.x * blockDim.x + threadIdx.x;
  if (t >= total) return;
  int b = t / S;
  const float* P = pts + (size_t)b * N * 3;
  float cx = centers[(size_t)t * 3 + 0];
  float cy = centers[(size_t)t * 3 + 1];
  float cz = centers[(size_t)t * 3 + 2];
  int cnt = 0, first = -1;
  for (int n = 0; n < N && cnt < K; ++n) {
    float dx = P[n * 3 + 0] - cx, dy = P[n * 3 + 1] - cy, dz = P[n * 3 + 2] - cz;
    if (dx * dx + dy * dy + dz * dz <= r2) {
      gi[(size_t)t * K + cnt] = n;
      if (first < 0) first = n;
      ++cnt;
    }
  }
  int fill = (first < 0) ? 0 : first;
  for (int j = cnt; j < K; ++j) gi[(size_t)t * K + j] = fill;
}

// Build grouped fp16 activation rows: [centered xyz | feats | zero-pad] (K padded to mult of 32)
__global__ void group_kernel(const float* __restrict__ xyz, const float* __restrict__ centers,
                             const _Float16* __restrict__ feats, const int* __restrict__ gi,
                             _Float16* __restrict__ out, int N, int SK, int K,
                             int Cf, int Cpad, int total) {
  int t = blockIdx.x * blockDim.x + threadIdx.x;
  if (t >= total) return;
  int b = t / SK;
  int sk = t % SK;
  int s = sk / K;
  int n = gi[t];
  int crow = b * (SK / K) + s;
  const float* C = centers + (size_t)crow * 3;
  const float* P = xyz + ((size_t)b * N + n) * 3;
  _Float16* o = out + (size_t)t * Cpad;
  o[0] = (_Float16)(P[0] - C[0]);
  o[1] = (_Float16)(P[1] - C[1]);
  o[2] = (_Float16)(P[2] - C[2]);
  if (feats) {
    const _Float16* F = feats + ((size_t)b * N + n) * Cf;
    for (int c = 0; c < Cf; ++c) o[3 + c] = F[c];
  }
  for (int c = 3 + Cf; c < Cpad; ++c) o[c] = (_Float16)0.0f;
}

// SA3 group_all: out[row] = [xyz | feats | pad]  (no centering, per reference)
__global__ void concat_all_kernel(const float* __restrict__ xyz, const _Float16* __restrict__ feats,
                                  _Float16* __restrict__ out, int Cf, int Cpad, int total) {
  int t = blockIdx.x * blockDim.x + threadIdx.x;
  if (t >= total) return;
  _Float16* o = out + (size_t)t * Cpad;
  o[0] = (_Float16)xyz[(size_t)t * 3 + 0];
  o[1] = (_Float16)xyz[(size_t)t * 3 + 1];
  o[2] = (_Float16)xyz[(size_t)t * 3 + 2];
  const _Float16* F = feats + (size_t)t * Cf;
  for (int c = 0; c < Cf; ++c) o[3 + c] = F[c];
  for (int c = 3 + Cf; c < Cpad; ++c) o[c] = (_Float16)0.0f;
}

// ---------------------------------------------------------------------------
// Weight prep: fold BN scale (g*INV_BN) into W and bias, convert to fp16, and
// swizzle into this kernel's contiguous-k 16-bit B (32x16 per chunk) per-lane layout:
//   Bsw[((ct*KC + kc)*32 + lane)*16 + e]  <->  B[k, n] = scale[n] * W[n, k]
//   with n = ct*16 + lane%16,  k = kc*32 + (lane/16)*16 + e
// so each lane's 16 halves are one contiguous 32-byte run (two b128 loads).
// ---------------------------------------------------------------------------
__global__ void prep_weights_kernel(const float* __restrict__ W, const float* __restrict__ Bb,
                                    const float* __restrict__ G, const float* __restrict__ Beta,
                                    _Float16* __restrict__ bsw, float* __restrict__ bias2,
                                    int Cin, int Cout, int Kpad, int Npad) {
  int t = blockIdx.x * blockDim.x + threadIdx.x;
  if (t < Npad) {
    float bv = 0.0f;
    if (t < Cout) {
      float sc = (G != nullptr) ? G[t] * INV_BN : 1.0f;
      bv = sc * Bb[t] + ((Beta != nullptr) ? Beta[t] : 0.0f);
    }
    bias2[t] = bv;
  }
  int totalW = Npad * Kpad;
  if (t >= totalW) return;
  int e    = t & 15;
  int lane = (t >> 4) & 31;
  int rem  = t >> 9;
  int KC   = Kpad >> 5;
  int kc   = rem % KC;
  int ct   = rem / KC;
  int n = ct * 16 + (lane & 15);
  int k = kc * 32 + ((lane >> 4) << 4) + e;
  float v = 0.0f;
  if (n < Cout && k < Cin) {
    float sc = (G != nullptr) ? G[n] * INV_BN : 1.0f;
    v = sc * W[(size_t)n * Cin + k];
  }
  bsw[t] = (_Float16)v;
}

// ---------------------------------------------------------------------------
// WMMA GEMM, register-blocked: one wave computes a 16 x (16*NB) output strip.
// Per K-chunk: issue ALL loads first (2x b128 A + NB x 2x b128 B, distinct
// registers) so up to 2+2*NB b128 loads are in flight, then run the NB WMMAs.
// This lets the scheduler overlap v_wmma with outstanding vmem instead of
// fully serializing on s_wait_loadcnt 0 per tile.
// ---------------------------------------------------------------------------
template <int NB>
__global__ void __launch_bounds__(32)
gemm_wmma_nb(const _Float16* __restrict__ A, const _Float16* __restrict__ Bsw,
             const float* __restrict__ bias, _Float16* __restrict__ outh,
             float* __restrict__ outf, int K, int N, int relu, int realN) {
  const int lane = threadIdx.x;
  const int rowTile = blockIdx.x;
  const int ctBase  = blockIdx.y * NB;
  const int KC = K >> 5;
  const int h  = lane >> 4;     // half-wave: selects K sub-range (A) / K offset (B)
  const int nl = lane & 15;
  // A layout (16-bit A, 16x32): lane row = lane%16; halves 0-15 <-> K=16h..16h+15
  // (matches B swizzle) -> one contiguous 32B run per chunk.
  const _Float16* Arow  = A + (size_t)(rowTile * 16 + nl) * K + 16 * h;
  const _Float16* Bbase = Bsw + (size_t)ctBase * KC * 512 + lane * 16;
  v8f acc[NB];
#pragma unroll
  for (int i = 0; i < NB; ++i) acc[i] = (v8f){};
  for (int kc = 0; kc < KC; ++kc) {
    // ---- load phase: A pair + NB B pairs, all into distinct registers ----
    v8h a0 = *(const v8h*)(Arow + kc * 32);
    v8h a1 = *(const v8h*)(Arow + kc * 32 + 8);
    v8h b0s[NB], b1s[NB];
#pragma unroll
    for (int nt = 0; nt < NB; ++nt) {
      const _Float16* Bp = Bbase + ((size_t)nt * KC + kc) * 512;
      b0s[nt] = *(const v8h*)(Bp);
      b1s[nt] = *(const v8h*)(Bp + 8);
    }
    if (kc + 1 < KC) __builtin_prefetch(Arow + (kc + 1) * 32, 0, 3);
    // ---- compute phase: NB WMMAs sharing the A registers ----
    v16h av = __builtin_shufflevector(a0, a1, 0,1,2,3,4,5,6,7,8,9,10,11,12,13,14,15);
#pragma unroll
    for (int nt = 0; nt < NB; ++nt) {
      v16h bv = __builtin_shufflevector(b0s[nt], b1s[nt],
                                        0,1,2,3,4,5,6,7,8,9,10,11,12,13,14,15);
      acc[nt] = __builtin_amdgcn_wmma_f32_16x16x32_f16(false, av, false, bv, (short)0,
                                                       acc[nt], false, false);
    }
  }
#pragma unroll
  for (int nt = 0; nt < NB; ++nt) {
    const int ncol = (ctBase + nt) * 16 + nl;   // C/D layout: lane column = lane%16
    const float bv = bias[ncol];
#pragma unroll
    for (int i = 0; i < 8; ++i) {               // C/D layout: row = vgpr + 8*(lane/16)
      float v = acc[nt][i] + bv;
      if (relu) v = fmaxf(v, 0.0f);
      int orow = rowTile * 16 + i + 8 * h;
      if (outh) {
        outh[(size_t)orow * N + ncol] = (_Float16)v;
      } else if (ncol < realN) {
        outf[(size_t)orow * realN + ncol] = v;
      }
    }
  }
}

// Max over group dimension: in [BS*K, C] -> out [BS, C]
__global__ void maxpool_kernel(const _Float16* __restrict__ in, _Float16* __restrict__ out,
                               int K, int C, int total) {
  int t = blockIdx.x * blockDim.x + threadIdx.x;
  if (t >= total) return;
  int bs = t / C, c = t % C;
  const _Float16* p = in + (size_t)bs * K * C + c;
  float m = -3.0e38f;
  for (int k = 0; k < K; ++k) m = fmaxf(m, (float)p[(size_t)k * C]);
  out[t] = (_Float16)m;
}

// ---------------------------------------------------------------------------
extern "C" void kernel_launch(void* const* d_in, const int* in_sizes, int n_in,
                              void* d_out, int out_size, void* d_ws, size_t ws_size,
                              hipStream_t stream) {
  (void)in_sizes; (void)n_in; (void)out_size; (void)ws_size;
  const float* points = (const float*)d_in[0];
  const int B = 16, N0 = 4096;
  const int S1 = 1024, K1 = 32;
  const int S2 = 128,  K2 = 64;

  struct LayerDef { int Cin, Cout, Kpad, Npad; const float *w, *b, *g, *be; };
  LayerDef L[12];
  const int cins[12]  = {3, 64, 64,   131, 128, 128,  259, 256, 512,  1024, 512, 256};
  const int couts[12] = {64, 64, 128, 128, 128, 256,  256, 512, 1024, 512,  256, 40};
  int pi = 1;
  for (int i = 0; i < 12; ++i) {
    L[i].Cin = cins[i]; L[i].Cout = couts[i];
    L[i].Kpad = (cins[i] + 31) & ~31;
    L[i].Npad = (couts[i] + 15) & ~15;
    L[i].w = (const float*)d_in[pi++];
    L[i].b = (const float*)d_in[pi++];
    if (i < 11) { L[i].g = (const float*)d_in[pi++]; L[i].be = (const float*)d_in[pi++]; }
    else        { L[i].g = nullptr;                  L[i].be = nullptr; }
  }

  // ---- workspace bump allocator ----
  char* ws = (char*)d_ws;
  size_t off = 0;
  auto alloc = [&](size_t bytes) -> void* {
    void* p = ws + off;
    off = (off + bytes + 255) & ~(size_t)255;
    return p;
  };
  _Float16* WB[12]; float* BIAS[12];
  for (int i = 0; i < 12; ++i) {
    WB[i]   = (_Float16*)alloc((size_t)L[i].Npad * L[i].Kpad * 2);
    BIAS[i] = (float*)   alloc((size_t)L[i].Npad * 4);
  }
  int*      fps1   = (int*)     alloc((size_t)B * S1 * 4);
  int*      fps2   = (int*)     alloc((size_t)B * S2 * 4);
  float*    l1_xyz = (float*)   alloc((size_t)B * S1 * 3 * 4);
  float*    l2_xyz = (float*)   alloc((size_t)B * S2 * 3 * 4);
  int*      gi1    = (int*)     alloc((size_t)B * S1 * K1 * 4);
  int*      gi2    = (int*)     alloc((size_t)B * S2 * K2 * 4);
  _Float16* l1_f   = (_Float16*)alloc((size_t)B * S1 * 128 * 2);
  _Float16* l2_f   = (_Float16*)alloc((size_t)B * S2 * 256 * 2);
  _Float16* x3     = (_Float16*)alloc((size_t)B * 1024 * 2);
  _Float16* Agrp   = (_Float16*)alloc((size_t)B * S2 * K2 * 160 * 2);   // 40 MB (max group buf)
  _Float16* bufA   = (_Float16*)alloc((size_t)B * S1 * K1 * 128 * 2);   // 128 MB
  _Float16* bufB   = (_Float16*)alloc((size_t)B * S1 * K1 * 64 * 2);    // 64 MB

  auto gemm = [&](const _Float16* A, int li, _Float16* outh, float* outf,
                  int rows, int relu, int realN) {
    int colTiles = L[li].Npad / 16;
    if ((colTiles & 3) == 0) {
      dim3 g((unsigned)(rows / 16), (unsigned)(colTiles / 4));
      gemm_wmma_nb<4><<<g, 32, 0, stream>>>(A, WB[li], BIAS[li], outh, outf,
                                            L[li].Kpad, L[li].Npad, relu, realN);
    } else {  // final FC: Npad=48 -> 3 column tiles
      dim3 g((unsigned)(rows / 16), (unsigned)(colTiles / 3));
      gemm_wmma_nb<3><<<g, 32, 0, stream>>>(A, WB[li], BIAS[li], outh, outf,
                                            L[li].Kpad, L[li].Npad, relu, realN);
    }
  };

  // ---- fold BN + swizzle all weights ----
  for (int i = 0; i < 12; ++i) {
    int tot = L[i].Npad * L[i].Kpad;
    prep_weights_kernel<<<(tot + 255) / 256, 256, 0, stream>>>(
        L[i].w, L[i].b, L[i].g, L[i].be, WB[i], BIAS[i],
        L[i].Cin, L[i].Cout, L[i].Kpad, L[i].Npad);
  }

  // ---- SA1: 4096 -> 1024 centers, K=32, MLP 3->64->64->128 ----
  fps_kernel<<<B, 256, 0, stream>>>(points, fps1, N0, S1);
  gather_xyz_kernel<<<(B * S1 + 255) / 256, 256, 0, stream>>>(points, fps1, l1_xyz, N0, S1, B * S1);
  ballquery_kernel<<<(B * S1 + 255) / 256, 256, 0, stream>>>(l1_xyz, points, gi1,
                                                             0.2f * 0.2f, N0, S1, K1, B * S1);
  group_kernel<<<(B * S1 * K1 + 255) / 256, 256, 0, stream>>>(points, l1_xyz, nullptr, gi1,
                                                              Agrp, N0, S1 * K1, K1, 0, 32,
                                                              B * S1 * K1);
  const int rows1 = B * S1 * K1;                    // 524288
  gemm(Agrp, 0, bufA, nullptr, rows1, 1, 0);
  gemm(bufA, 1, bufB, nullptr, rows1, 1, 0);
  gemm(bufB, 2, bufA, nullptr, rows1, 1, 0);
  maxpool_kernel<<<(B * S1 * 128 + 255) / 256, 256, 0, stream>>>(bufA, l1_f, K1, 128, B * S1 * 128);

  // ---- SA2: 1024 -> 128 centers, K=64, MLP 131->128->128->256 ----
  fps_kernel<<<B, 256, 0, stream>>>(l1_xyz, fps2, S1, S2);
  gather_xyz_kernel<<<(B * S2 + 255) / 256, 256, 0, stream>>>(l1_xyz, fps2, l2_xyz, S1, S2, B * S2);
  ballquery_kernel<<<(B * S2 + 255) / 256, 256, 0, stream>>>(l2_xyz, l1_xyz, gi2,
                                                             0.4f * 0.4f, S1, S2, K2, B * S2);
  group_kernel<<<(B * S2 * K2 + 255) / 256, 256, 0, stream>>>(l1_xyz, l2_xyz, l1_f, gi2,
                                                              Agrp, S1, S2 * K2, K2, 128, 160,
                                                              B * S2 * K2);
  const int rows2 = B * S2 * K2;                    // 131072
  gemm(Agrp, 3, bufA, nullptr, rows2, 1, 0);
  gemm(bufA, 4, bufB, nullptr, rows2, 1, 0);
  gemm(bufB, 5, bufA, nullptr, rows2, 1, 0);
  maxpool_kernel<<<(B * S2 * 256 + 255) / 256, 256, 0, stream>>>(bufA, l2_f, K2, 256, B * S2 * 256);

  // ---- SA3 (group_all): 128 rows/batch, MLP 259->256->512->1024 ----
  concat_all_kernel<<<(B * S2 + 255) / 256, 256, 0, stream>>>(l2_xyz, l2_f, Agrp, 256, 288, B * S2);
  const int rows3 = B * S2;                         // 2048
  gemm(Agrp, 6, bufA, nullptr, rows3, 1, 0);
  gemm(bufA, 7, bufB, nullptr, rows3, 1, 0);
  gemm(bufB, 8, bufA, nullptr, rows3, 1, 0);
  maxpool_kernel<<<(B * 1024 + 255) / 256, 256, 0, stream>>>(bufA, x3, S2, 1024, B * 1024);

  // ---- Head: 1024->512->256->40 (last layer fp32 logits straight to d_out) ----
  gemm(x3,   9, bufA, nullptr, B, 1, 0);
  gemm(bufA, 10, bufB, nullptr, B, 1, 0);
  gemm(bufB, 11, nullptr, (float*)d_out, B, 0, 40);
}
